// CSAB3_1322849927417
// MI455X (gfx1250) — matrix operand order repaired and checked by reading, not compiled, not run on hardware
//
#include <hip/hip_runtime.h>
#include <hip/hip_bf16.h>

typedef _Float16 half_t;
typedef __attribute__((ext_vector_type(16))) _Float16 v16h;
typedef __attribute__((ext_vector_type(8)))  _Float16 v8h;
typedef __attribute__((ext_vector_type(8)))  float    v8f;

union HV16 { v16h v; unsigned u[8]; v8h h[2]; };

constexpr int Bsz   = 8;
constexpr int Nset  = 1024;
constexpr int Dm    = 512;
constexpr int Hh    = 8;
constexpr int DH    = 64;            // Dm / Hh
constexpr int Mrows = Bsz * Nset;    // 8192

// ---------------- workspace layout (units: half_t elements) ----------------
constexpr size_t OFF_XH  = 0;
constexpr size_t OFF_YH  = OFF_XH  + (size_t)Mrows * Dm;
constexpr size_t OFF_WT  = OFF_YH  + (size_t)Mrows * Dm;          // 12 x [Dm][Dm] (transposed)
constexpr size_t OFF_WTX = OFF_WT  + 12ull * Dm * Dm;             // [Dm][2*Dm]
constexpr size_t OFF_WTY = OFF_WTX + (size_t)Dm * 2 * Dm;
constexpr size_t OFF_QKV = OFF_WTY + (size_t)Dm * 2 * Dm;         // 12 x [Mrows][Dm]
constexpr size_t OFF_O   = OFF_QKV + 12ull * Mrows * Dm;          // 4 x [Mrows][Dm]

// ---------------- job descriptors (passed by value as kernargs) ------------
struct TransJob  { const float* W; half_t* Wt; int din; };
struct TransJobs { TransJob j[14]; };
struct GemmJob   { const half_t* A; const half_t* Wt; const float* bias; half_t* C; };
struct GemmJobs  { GemmJob j[12]; };
struct AttnJob   { const half_t* Q; const half_t* K; const half_t* V; half_t* O; };
struct AttnJobs  { AttnJob j[4]; };
struct FinalJob  { const half_t* A0; const half_t* A1; const half_t* Wt;
                   const float* bias; const float* residual; float* out; };
struct FinalJobs { FinalJob j[2]; };

// ---------------- Tensor Data Mover support --------------------------------
#if defined(__has_builtin)
#  if __has_builtin(__builtin_amdgcn_tensor_load_to_lds) && \
      __has_builtin(__builtin_amdgcn_s_wait_tensorcnt)
#    define HAVE_TDM 1
#  endif
#endif
#ifndef HAVE_TDM
#  define HAVE_TDM 0
#endif

typedef __attribute__((ext_vector_type(4))) unsigned tdm_v4u;
typedef __attribute__((ext_vector_type(8))) int      tdm_v8i;
typedef __attribute__((ext_vector_type(4))) int      tdm_v4i;

// Low 32 bits of the LDS generic address are the LDS byte offset (ISA 10.2).
__device__ __forceinline__ unsigned lds_off(const void* p) {
    return (unsigned)(uintptr_t)p;
}

#if HAVE_TDM
// Issue a TDM 2-D f16 tile load global->LDS. Tile rows are tile_d0 halves
// (=64B), LDS rows padded by 4 dwords every 16 dwords -> 80B LDS row stride,
// matching half_t rows of 40 elements. D# bitfields per CDNA5 ISA ch.8.
__device__ __forceinline__ void tdm_load_tile(unsigned lds_addr,
                                              const half_t* gaddr,
                                              int tile_d0, int tile_d1,
                                              int row_stride_elems) {
    uint64_t ga = (uint64_t)(uintptr_t)gaddr;
    tdm_v4u g0;
    g0[0] = 1u;                                              // count=1 (valid)
    g0[1] = lds_addr;                                        // lds_addr[31:0]
    g0[2] = (unsigned)ga;                                    // global_addr lo
    g0[3] = (unsigned)((ga >> 32) & 0x01FFFFFFu) | 0x80000000u; // ga hi | type=2
    tdm_v8i g1;
    // data_size=1 (2B), pad_enable, pad_interval=3 (16 dwords), pad_amount=3 (4 dwords)
    g1[0] = (1 << 16) | (1 << 20) | (3 << 22) | (3 << 25);
    const int td0 = row_stride_elems;                        // tensor_dim0
    const int td1 = tile_d1;                                 // tensor_dim1
    g1[1] = (td0 & 0xFFFF) << 16;                            // tensor_dim0[15:0]
    g1[2] = ((td0 >> 16) & 0xFFFF) | ((td1 & 0xFFFF) << 16);
    g1[3] = ((td1 >> 16) & 0xFFFF) | (tile_d0 << 16);        // tile_dim0
    g1[4] = tile_d1;                                         // tile_dim1, tile_dim2=0
    g1[5] = row_stride_elems;                                // tensor_dim0_stride lo
    g1[6] = 0;
    g1[7] = 0;
    tdm_v4i z4 = (tdm_v4i)0;
#if __clang_major__ >= 23
    tdm_v8i z8 = (tdm_v8i)0;
    __builtin_amdgcn_tensor_load_to_lds(g0, g1, z4, z4, z8, 0);
#else
    __builtin_amdgcn_tensor_load_to_lds(g0, g1, z4, z4, 0);
#endif
}

// s_wait_tensorcnt needs a constant immediate; select encoding with a branch.
__device__ __forceinline__ void tdm_wait(bool one_outstanding) {
    if (one_outstanding) __builtin_amdgcn_s_wait_tensorcnt(1);
    else                 __builtin_amdgcn_s_wait_tensorcnt(0);
}
#endif

// ---------------- kernel 0: f32 -> f16 convert -----------------------------
__global__ __launch_bounds__(256) void convert_f32_f16(const float* __restrict__ in,
                                                       half_t* __restrict__ out, int n) {
    for (int i = blockIdx.x * blockDim.x + threadIdx.x; i < n; i += gridDim.x * blockDim.x)
        out[i] = (half_t)in[i];
}

// ---------------- kernel 1: weight transpose f32[din][512] -> f16[512][din] -
__global__ __launch_bounds__(256) void transpose_weights(TransJobs jobs) {
    const TransJob jb = jobs.j[blockIdx.z];
    const int total = jb.din * Dm;
    for (int i = blockIdx.x * blockDim.x + threadIdx.x; i < total; i += gridDim.x * blockDim.x) {
        int r = i / Dm, c = i - r * Dm;             // W[r][c]
        jb.Wt[(size_t)c * jb.din + r] = (half_t)jb.W[i];
    }
}

// ---------------- shared GEMM mainloop body (fragments + 4 WMMA) -----------
__device__ __forceinline__ void gemm_step(const half_t (*As)[40], const half_t (*Bs)[40],
                                          int w, int lm, int lh, v8f acc[4]) {
    HV16 a;
#pragma unroll
    for (int j = 0; j < 8; ++j) {                 // A-layout: K = e + 16*half
        int kk = 2 * j + lh * 16;
        a.u[j] = *(const unsigned*)&As[w * 16 + lm][kk];
    }
#pragma unroll
    for (int nb = 0; nb < 4; ++nb) {
        HV16 b;
        b.v = *(const v16h*)&Bs[nb * 16 + lm][lh * 16];   // K = e + 16*half
        acc[nb] = __builtin_amdgcn_wmma_f32_16x16x32_f16(
            false, a.v, false, b.v, (short)0, acc[nb], false, false);
    }
}

// ---------------- kernel 2: QKV GEMM  C[M,512] = A[M,512] * W + b ----------
// WG = 256 threads (8 waves). WG tile 128(M) x 64(N). Wave tile 16 x 64.
// TDM double-buffers the K-tiles: wave0 DMAs A, wave1 DMAs B for step k+1
// while all 8 waves run WMMA on step k.
__global__ __launch_bounds__(256) void gemm_qkv(GemmJobs jobs) {
    const GemmJob jb = jobs.j[blockIdx.z];
    __shared__ __attribute__((aligned(16))) half_t As[2][128][40];
    __shared__ __attribute__((aligned(16))) half_t Bs[2][64][40];
    const int tid = threadIdx.x;
    const int w = tid >> 5, lane = tid & 31, lm = lane & 15, lh = lane >> 4;
    const int m0 = blockIdx.y * 128;
    const int n0 = blockIdx.x * 64;

    v8f acc[4] = {};
    const int nk = Dm / 32;

#if HAVE_TDM
    if (w == 0) tdm_load_tile(lds_off(&As[0][0][0]), jb.A  + (size_t)m0 * Dm, 32, 128, Dm);
    if (w == 1) tdm_load_tile(lds_off(&Bs[0][0][0]), jb.Wt + (size_t)n0 * Dm, 32,  64, Dm);
    for (int it = 0; it < nk; ++it) {
        const int buf = it & 1;
        __syncthreads();                       // prior readers of buf^1 done
        if (it + 1 < nk) {
            const int k1 = (it + 1) * 32, b1 = (it + 1) & 1;
            if (w == 0) tdm_load_tile(lds_off(&As[b1][0][0]), jb.A  + (size_t)m0 * Dm + k1, 32, 128, Dm);
            if (w == 1) tdm_load_tile(lds_off(&Bs[b1][0][0]), jb.Wt + (size_t)n0 * Dm + k1, 32,  64, Dm);
        }
        if (w < 2) tdm_wait(it + 1 < nk);
        __syncthreads();                       // tile `it` now visible to WG
        gemm_step(As[buf], Bs[buf], w, lm, lh, acc);
    }
#else
    for (int it = 0; it < nk; ++it) {
        const int k0 = it * 32;
        {
            int row = tid >> 1, seg = tid & 1;
            const half_t* gp = jb.A + (size_t)(m0 + row) * Dm + k0 + seg * 16;
            *(v8h*)&As[0][row][seg * 16]     = *(const v8h*)gp;
            *(v8h*)&As[0][row][seg * 16 + 8] = *(const v8h*)(gp + 8);
        }
        {
            int row = tid >> 2, seg = tid & 3;
            *(v8h*)&Bs[0][row][seg * 8] =
                *(const v8h*)(jb.Wt + (size_t)(n0 + row) * Dm + k0 + seg * 8);
        }
        __syncthreads();
        gemm_step(As[0], Bs[0], w, lm, lh, acc);
        __syncthreads();
    }
#endif

    // epilogue: +bias, store f16 (C layout: m = r + 8*half, n = lane%16)
#pragma unroll
    for (int nb = 0; nb < 4; ++nb) {
        int n = n0 + nb * 16 + lm;
        float bval = jb.bias[n];
#pragma unroll
        for (int r = 0; r < 8; ++r) {
            int m = m0 + w * 16 + r + 8 * lh;
            jb.C[(size_t)m * Dm + n] = (half_t)(acc[nb][r] + bval);
        }
    }
}

// ---------------- kernel 3: flash attention with Q-residual ----------------
// grid: (qtiles=8, B*H=64, branch=4). WG = 8 waves, each wave 16 query rows.
__global__ __launch_bounds__(256) void attention(AttnJobs jobs) {
    const AttnJob jb = jobs.j[blockIdx.z];
    __shared__ __attribute__((aligned(16))) half_t Ks[32][72];    // [kv][d64 + pad]
    __shared__ __attribute__((aligned(16))) half_t Vs[64][40];    // transposed: [d][kv32 + pad]
    __shared__ __attribute__((aligned(16))) half_t Ps[8][16][40]; // per-wave P tile
    const int tid = threadIdx.x;
    const int w = tid >> 5, lane = tid & 31, lm = lane & 15, lh = lane >> 4;
    const int bh = blockIdx.y;
    const int bb = bh >> 3, h = bh & 7;
    const int fb = h * DH;
    const int rowbase = bb * Nset;
    const int qg = rowbase + blockIdx.x * 128 + w * 16;
    const float scale = 0.04419417382415922f;   // 1/sqrt(512)

    // resident Q fragments (two K=32 chunks covering d_h = 64)
    HV16 qa[2];
#pragma unroll
    for (int c = 0; c < 2; ++c)
#pragma unroll
        for (int j = 0; j < 8; ++j) {
            int kk = c * 32 + 2 * j + lh * 16;
            qa[c].u[j] = *(const unsigned*)(jb.Q + (size_t)(qg + lm) * Dm + fb + kk);
        }

    v8f o[4] = {};
    float mrow[8], lrow[8];
#pragma unroll
    for (int r = 0; r < 8; ++r) { mrow[r] = -1e30f; lrow[r] = 0.f; }

    for (int kb = 0; kb < Nset / 32; ++kb) {
        const int kvg = rowbase + kb * 32;
        __syncthreads();
        { // stage K tile (copy, K-dim contiguous)
            int kvl = tid >> 3, seg = tid & 7;
            *(v8h*)&Ks[kvl][seg * 8] =
                *(const v8h*)(jb.K + (size_t)(kvg + kvl) * Dm + fb + seg * 8);
        }
        { // stage V tile transposed
            int kvl = tid & 31, d0 = (tid >> 5) * 8;
            v8h vv = *(const v8h*)(jb.V + (size_t)(kvg + kvl) * Dm + fb + d0);
#pragma unroll
            for (int i = 0; i < 8; ++i) Vs[d0 + i][kvl] = vv[i];
        }
        __syncthreads();

        // S = Q * K^T  (16 x 32, two 16x16 accumulators)
        v8f s0 = {}, s1 = {};
        {
            HV16 b00, b01, b10, b11;
            b00.v = *(const v16h*)&Ks[lm][lh * 16];
            b01.v = *(const v16h*)&Ks[lm][32 + lh * 16];
            b10.v = *(const v16h*)&Ks[16 + lm][lh * 16];
            b11.v = *(const v16h*)&Ks[16 + lm][32 + lh * 16];
            s0 = __builtin_amdgcn_wmma_f32_16x16x32_f16(false, qa[0].v, false, b00.v, (short)0, s0, false, false);
            s0 = __builtin_amdgcn_wmma_f32_16x16x32_f16(false, qa[1].v, false, b01.v, (short)0, s0, false, false);
            s1 = __builtin_amdgcn_wmma_f32_16x16x32_f16(false, qa[0].v, false, b10.v, (short)0, s1, false, false);
            s1 = __builtin_amdgcn_wmma_f32_16x16x32_f16(false, qa[1].v, false, b11.v, (short)0, s1, false, false);
        }

        // online softmax update (reductions across the 16-lane half groups)
#pragma unroll
        for (int r = 0; r < 8; ++r) {
            float v0 = s0[r] * scale, v1 = s1[r] * scale;
            float bmax = fmaxf(v0, v1);
#pragma unroll
            for (int d = 1; d < 16; d <<= 1) bmax = fmaxf(bmax, __shfl_xor(bmax, d, 32));
            float mn = fmaxf(mrow[r], bmax);
            float alpha = __expf(mrow[r] - mn);
            float p0 = __expf(v0 - mn), p1 = __expf(v1 - mn);
            float ps = p0 + p1;
#pragma unroll
            for (int d = 1; d < 16; d <<= 1) ps += __shfl_xor(ps, d, 32);
            lrow[r] = lrow[r] * alpha + ps;
            mrow[r] = mn;
#pragma unroll
            for (int nb = 0; nb < 4; ++nb) o[nb][r] *= alpha;
            int mloc = r + 8 * lh;
            Ps[w][mloc][lm]      = (half_t)p0;
            Ps[w][mloc][16 + lm] = (half_t)p1;
        }
        // wave-private LDS RAW: only this wave touches Ps[w]
        asm volatile("s_wait_dscnt 0" ::: "memory");

        // O += P * V
        HV16 pa;
#pragma unroll
        for (int j = 0; j < 8; ++j) {
            int kk = 2 * j + lh * 16;
            pa.u[j] = *(const unsigned*)&Ps[w][lm][kk];
        }
#pragma unroll
        for (int nb = 0; nb < 4; ++nb) {
            HV16 vb;
            vb.v = *(const v16h*)&Vs[nb * 16 + lm][lh * 16];
            o[nb] = __builtin_amdgcn_wmma_f32_16x16x32_f16(
                false, pa.v, false, vb.v, (short)0, o[nb], false, false);
        }
    }

    // epilogue: normalize, add Q residual, store f16
#pragma unroll
    for (int r = 0; r < 8; ++r) {
        float inv = 1.0f / lrow[r];
        int m = qg + r + 8 * lh;
#pragma unroll
        for (int nb = 0; nb < 4; ++nb) {
            int f = fb + nb * 16 + lm;
            float qres = (float)jb.Q[(size_t)m * Dm + f];
            jb.O[(size_t)m * Dm + f] = (half_t)(qres + o[nb][r] * inv);
        }
    }
}

// ---------------- kernel 4: out = res + relu(concat(A0,A1) @ W + b) --------
__global__ __launch_bounds__(256) void final_linear(FinalJobs jobs) {
    const FinalJob jb = jobs.j[blockIdx.z];
    __shared__ __attribute__((aligned(16))) half_t As[2][128][40];
    __shared__ __attribute__((aligned(16))) half_t Bs[2][64][40];
    const int tid = threadIdx.x;
    const int w = tid >> 5, lane = tid & 31, lm = lane & 15, lh = lane >> 4;
    const int m0 = blockIdx.y * 128;
    const int n0 = blockIdx.x * 64;

    v8f acc[4] = {};
    const int nk = (2 * Dm) / 32;

#if HAVE_TDM
    if (w == 0) tdm_load_tile(lds_off(&As[0][0][0]), jb.A0 + (size_t)m0 * Dm, 32, 128, Dm);
    if (w == 1) tdm_load_tile(lds_off(&Bs[0][0][0]), jb.Wt + (size_t)n0 * (2 * Dm), 32, 64, 2 * Dm);
    for (int it = 0; it < nk; ++it) {
        const int buf = it & 1;
        __syncthreads();
        if (it + 1 < nk) {
            const int k1 = (it + 1) * 32, b1 = (it + 1) & 1;
            const half_t* Asrc = (k1 < Dm) ? jb.A0 : jb.A1;
            const int kc = k1 & (Dm - 1);
            if (w == 0) tdm_load_tile(lds_off(&As[b1][0][0]), Asrc + (size_t)m0 * Dm + kc, 32, 128, Dm);
            if (w == 1) tdm_load_tile(lds_off(&Bs[b1][0][0]), jb.Wt + (size_t)n0 * (2 * Dm) + k1, 32, 64, 2 * Dm);
        }
        if (w < 2) tdm_wait(it + 1 < nk);
        __syncthreads();
        gemm_step(As[buf], Bs[buf], w, lm, lh, acc);
    }
#else
    for (int it = 0; it < nk; ++it) {
        const int k0 = it * 32;
        const half_t* Asrc = (k0 < Dm) ? jb.A0 : jb.A1;
        const int kc = k0 & (Dm - 1);
        {
            int row = tid >> 1, seg = tid & 1;
            const half_t* gp = Asrc + (size_t)(m0 + row) * Dm + kc + seg * 16;
            *(v8h*)&As[0][row][seg * 16]     = *(const v8h*)gp;
            *(v8h*)&As[0][row][seg * 16 + 8] = *(const v8h*)(gp + 8);
        }
        {
            int row = tid >> 2, seg = tid & 3;
            *(v8h*)&Bs[0][row][seg * 8] =
                *(const v8h*)(jb.Wt + (size_t)(n0 + row) * (2 * Dm) + k0 + seg * 8);
        }
        __syncthreads();
        gemm_step(As[0], Bs[0], w, lm, lh, acc);
        __syncthreads();
    }
#endif

#pragma unroll
    for (int nb = 0; nb < 4; ++nb) {
        int n = n0 + nb * 16 + lm;
        float bval = jb.bias[n];
#pragma unroll
        for (int r = 0; r < 8; ++r) {
            int m = m0 + w * 16 + r + 8 * lh;
            float v = fmaxf(acc[nb][r] + bval, 0.0f);
            jb.out[(size_t)m * Dm + n] = jb.residual[(size_t)m * Dm + n] + v;
        }
    }
}

// ---------------- host-side orchestration ----------------------------------
extern "C" void kernel_launch(void* const* d_in, const int* in_sizes, int n_in,
                              void* d_out, int out_size, void* d_ws, size_t ws_size,
                              hipStream_t stream) {
    half_t* ws = (half_t*)d_ws;
    const float* X = (const float*)d_in[0];
    const float* Y = (const float*)d_in[1];

    // 1) convert X, Y to f16
    convert_f32_f16<<<512, 256, 0, stream>>>(X, ws + OFF_XH, Mrows * Dm);
    convert_f32_f16<<<512, 256, 0, stream>>>(Y, ws + OFF_YH, Mrows * Dm);

    // 2) transpose + convert all weights: Wt[dout][din] f16
    TransJobs tj;
    for (int i = 0; i < 12; ++i)
        tj.j[i] = { (const float*)d_in[2 + 2 * i], ws + OFF_WT + (size_t)i * Dm * Dm, Dm };
    tj.j[12] = { (const float*)d_in[26], ws + OFF_WTX, 2 * Dm };
    tj.j[13] = { (const float*)d_in[28], ws + OFF_WTY, 2 * Dm };
    transpose_weights<<<dim3(256, 1, 14), 256, 0, stream>>>(tj);

    // 3) 12 QKV projections in one launch
    // branch order: q_xx k_xx v_xx q_xy k_xy v_xy q_yx k_yx v_yx q_yy k_yy v_yy
    const int srcX[12] = {1, 1, 1, 1, 0, 0, 0, 1, 1, 0, 0, 0};
    GemmJobs gj;
    for (int i = 0; i < 12; ++i) {
        gj.j[i].A    = srcX[i] ? (ws + OFF_XH) : (ws + OFF_YH);
        gj.j[i].Wt   = ws + OFF_WT + (size_t)i * Dm * Dm;
        gj.j[i].bias = (const float*)d_in[3 + 2 * i];
        gj.j[i].C    = ws + OFF_QKV + (size_t)i * Mrows * Dm;
    }
    gemm_qkv<<<dim3(Dm / 64, Mrows / 128, 12), 256, 0, stream>>>(gj);

    // 4) four attentions in one launch
    AttnJobs aj;
    for (int z = 0; z < 4; ++z) {
        aj.j[z].Q = ws + OFF_QKV + (size_t)(3 * z + 0) * Mrows * Dm;
        aj.j[z].K = ws + OFF_QKV + (size_t)(3 * z + 1) * Mrows * Dm;
        aj.j[z].V = ws + OFF_QKV + (size_t)(3 * z + 2) * Mrows * Dm;
        aj.j[z].O = ws + OFF_O   + (size_t)z * Mrows * Dm;
    }
    attention<<<dim3(Nset / 128, Bsz * Hh, 4), 256, 0, stream>>>(aj);

    // 5) output linears (relu + residual), f32 into d_out (out_x then out_y)
    FinalJobs fj;
    fj.j[0] = { ws + OFF_O + 0,                      ws + OFF_O + 1ull * Mrows * Dm,
                ws + OFF_WTX, (const float*)d_in[27], X, (float*)d_out };
    fj.j[1] = { ws + OFF_O + 2ull * Mrows * Dm,      ws + OFF_O + 3ull * Mrows * Dm,
                ws + OFF_WTY, (const float*)d_in[29], Y, (float*)d_out + (size_t)Mrows * Dm };
    final_linear<<<dim3(Dm / 64, Mrows / 128, 2), 256, 0, stream>>>(fj);
}
